// Attention_65377992180030
// MI455X (gfx1250) — matrix operand order repaired and checked by reading, not compiled, not run on hardware
//
#include <hip/hip_runtime.h>
#include <hip/hip_bf16.h>
#include <math.h>

#define HH 512
#define LL 4096
#define BB 64

typedef __attribute__((ext_vector_type(16))) __bf16 v16bf;
typedef __attribute__((ext_vector_type(8)))  float  v8f;

struct Pack32 { uint4 a, b; };   // 32 bytes = one v16bf fragment

static __device__ __forceinline__ unsigned short f2bf_raw(float f) {
  return __builtin_bit_cast(unsigned short, (__bf16)f);
}
static __device__ __forceinline__ unsigned int pack2(float lo, float hi) {
  return (unsigned int)f2bf_raw(lo) | ((unsigned int)f2bf_raw(hi) << 16);
}

// ---------------------------------------------------------------------------
// Kernel 1: qsum[b,g] = query[b,:] . Wa_w[g,:] + Wa_b[g] + Ua_b[g]
// ---------------------------------------------------------------------------
__global__ __launch_bounds__(HH) void qproj_kernel(const float* __restrict__ query,
                                                   const float* __restrict__ Wa_w,
                                                   const float* __restrict__ Wa_b,
                                                   const float* __restrict__ Ua_b,
                                                   float* __restrict__ qsum) {
  const int b = blockIdx.x;
  const int g = threadIdx.x;
  __shared__ float q[HH];
  q[g] = query[b * HH + g];
  __syncthreads();
  const float4* wr = (const float4*)(Wa_w + (size_t)g * HH);
  float acc = 0.f;
  for (int h = 0; h < HH / 4; ++h) {
    float4 w = wr[h];
    acc += q[4*h+0]*w.x + q[4*h+1]*w.y + q[4*h+2]*w.z + q[4*h+3]*w.w;
  }
  qsum[b * HH + g] = acc + Wa_b[g] + Ua_b[g];
}

// ---------------------------------------------------------------------------
// Kernel 2: Ua_w f32 -> bf16 (row-major [g][h]), L2-resident reusable operand
// ---------------------------------------------------------------------------
__global__ __launch_bounds__(256) void cvt_bf16_kernel(const float* __restrict__ src,
                                                       unsigned short* __restrict__ dst,
                                                       int n) {
  const int i = blockIdx.x * 256 + threadIdx.x;
  if (i < n) dst[i] = f2bf_raw(src[i]);
}

// ---------------------------------------------------------------------------
// Kernel 3: scores[b,l] = sum_g tanh(kp[b,l,g] + qsum[b,g]) * Va_w[g]
// kp computed on the fly with v_wmma_f32_16x16x32_bf16.
// Block = 128 threads (4 waves); block tile = (b, 64 rows of l).
// ---------------------------------------------------------------------------
static __device__ __forceinline__ void score_epilogue(const v8f& a, float q, float va,
                                                      float* sacc) {
#pragma unroll
  for (int rr = 0; rr < 8; ++rr) sacc[rr] += tanhf(a[rr] + q) * va;
}

__global__ __launch_bounds__(128) void
bahdanau_scores_kernel(const float* __restrict__ keys,
                       const unsigned short* __restrict__ Uab,
                       const float* __restrict__ qsum,
                       const float* __restrict__ Va_w,
                       const float* __restrict__ Va_b,
                       float* __restrict__ scores) {
  const int b  = blockIdx.x;
  const int l0 = blockIdx.y * 64;

  // keys tile as bf16, 16-byte blocks XOR-swizzled by row&15 (bank-conflict-free
  // A-fragment reads: lanes 0-15 cover 16 distinct bank groups).
  __shared__ unsigned short sk[64 * 512];   // 64 KB

  const int tid = threadIdx.x;

  // ---- stage keys[b, l0:l0+64, :] -> LDS bf16 (16B blocks, swizzled) ----
  {
    const float4* src = (const float4*)(keys + ((size_t)b * LL + l0) * HH);
    for (int i = tid; i < 64 * 64; i += 128) {   // i = row*64 + 16B-block
      const int row = i >> 6;
      const int cb  = i & 63;
      float4 v0 = src[i * 2 + 0];
      float4 v1 = src[i * 2 + 1];
      uint4 o;
      o.x = pack2(v0.x, v0.y);
      o.y = pack2(v0.z, v0.w);
      o.z = pack2(v1.x, v1.y);
      o.w = pack2(v1.z, v1.w);
      const int pcb = cb ^ (row & 15);
      *(uint4*)(sk + row * 512 + pcb * 8) = o;
    }
  }
  __syncthreads();

  const int wv   = tid >> 5;     // wave 0..3 -> l rows [16*wv, 16*wv+16)
  const int lane = tid & 31;
  const int r16  = lane & 15;
  const int hi   = lane >> 4;    // ISA half-wave selector

  const unsigned short* abase = sk + (wv * 16 + r16) * 512;

  float sacc[8] = {};

#pragma unroll 1
  for (int nc = 0; nc < 8; ++nc) {           // 4 g-tiles (N=16) per chunk
    v8f acc0 = {}, acc1 = {}, acc2 = {}, acc3 = {};
    // B layout (32x16 bf16): lanes 0-15 hold K=k0..k0+15 of column r16,
    // lanes 16-31 hold K=k0+16..k0+31 -> 32 contiguous bytes of Ua_bf16 row.
    const unsigned short* bp0 = Uab + ((size_t)((nc * 4 + 0) * 16 + r16)) * HH + hi * 16;
    const unsigned short* bp1 = bp0 + 16 * HH;
    const unsigned short* bp2 = bp0 + 32 * HH;
    const unsigned short* bp3 = bp0 + 48 * HH;

#pragma unroll 4
    for (int k = 0; k < 16; ++k) {
      const int k0 = 32 * k;
      // A layout (16x32 bf16): lanes 0-15: K k0..k0+15;
      // lanes 16-31: K k0+16..k0+31 (matches B).
      const int cb1 = (4 * k + 2 * hi)     ^ r16;
      const int cb2 = (4 * k + 2 * hi + 1) ^ r16;
      Pack32 pa;
      pa.a = *(const uint4*)(abase + cb1 * 8);
      pa.b = *(const uint4*)(abase + cb2 * 8);
      const v16bf af = __builtin_bit_cast(v16bf, pa);

      Pack32 pb;
      pb.a = *(const uint4*)(bp0 + k0);
      pb.b = *(const uint4*)(bp0 + k0 + 8);
      acc0 = __builtin_amdgcn_wmma_f32_16x16x32_bf16(false, af, false,
             __builtin_bit_cast(v16bf, pb), (short)0, acc0, false, false);
      pb.a = *(const uint4*)(bp1 + k0);
      pb.b = *(const uint4*)(bp1 + k0 + 8);
      acc1 = __builtin_amdgcn_wmma_f32_16x16x32_bf16(false, af, false,
             __builtin_bit_cast(v16bf, pb), (short)0, acc1, false, false);
      pb.a = *(const uint4*)(bp2 + k0);
      pb.b = *(const uint4*)(bp2 + k0 + 8);
      acc2 = __builtin_amdgcn_wmma_f32_16x16x32_bf16(false, af, false,
             __builtin_bit_cast(v16bf, pb), (short)0, acc2, false, false);
      pb.a = *(const uint4*)(bp3 + k0);
      pb.b = *(const uint4*)(bp3 + k0 + 8);
      acc3 = __builtin_amdgcn_wmma_f32_16x16x32_bf16(false, af, false,
             __builtin_bit_cast(v16bf, pb), (short)0, acc3, false, false);
    }

    // C layout: lane holds column g = g0 + (lane&15); VGPR r holds row r + 8*hi.
    {
      const int g0 = nc * 64 + r16;
      score_epilogue(acc0, qsum[b * HH + g0 +  0], Va_w[g0 +  0], sacc);
      score_epilogue(acc1, qsum[b * HH + g0 + 16], Va_w[g0 + 16], sacc);
      score_epilogue(acc2, qsum[b * HH + g0 + 32], Va_w[g0 + 32], sacc);
      score_epilogue(acc3, qsum[b * HH + g0 + 48], Va_w[g0 + 48], sacc);
    }
  }

  // Reduce over the 16 columns held by each half-wave (xor 1,2,4,8 stays in-half).
  const float vb = Va_b[0];
#pragma unroll
  for (int rr = 0; rr < 8; ++rr) {
    float s = sacc[rr];
    s += __shfl_xor(s, 1, 32);
    s += __shfl_xor(s, 2, 32);
    s += __shfl_xor(s, 4, 32);
    s += __shfl_xor(s, 8, 32);
    sacc[rr] = s;
  }
  if (r16 == 0) {  // lane 0 -> rows 0..7 of the wave tile, lane 16 -> rows 8..15
    const int mbase = l0 + wv * 16 + hi * 8;
#pragma unroll
    for (int rr = 0; rr < 8; ++rr)
      scores[(size_t)b * LL + mbase + rr] = sacc[rr] + vb;
  }
}

// ---------------------------------------------------------------------------
// Kernel 4: masked softmax over L per batch
// ---------------------------------------------------------------------------
__global__ __launch_bounds__(256) void softmax_kernel(const float* __restrict__ scores,
                                                      const unsigned char* __restrict__ mask,
                                                      float* __restrict__ attn) {
  const int b = blockIdx.x;
  const int tid = threadIdx.x;
  __shared__ float red[8];
  float vals[LL / 256];
  float mx = -INFINITY;
#pragma unroll
  for (int i = 0; i < LL / 256; ++i) {
    const int l = i * 256 + tid;
    float s = scores[(size_t)b * LL + l];
    s = mask[(size_t)b * LL + l] ? s : -INFINITY;
    vals[i] = s;
    mx = fmaxf(mx, s);
  }
  for (int m = 16; m >= 1; m >>= 1) mx = fmaxf(mx, __shfl_xor(mx, m, 32));
  if ((tid & 31) == 0) red[tid >> 5] = mx;
  __syncthreads();
  float bm = red[0];
#pragma unroll
  for (int w = 1; w < 8; ++w) bm = fmaxf(bm, red[w]);
  __syncthreads();
  float sum = 0.f;
#pragma unroll
  for (int i = 0; i < LL / 256; ++i) { vals[i] = __expf(vals[i] - bm); sum += vals[i]; }
  for (int m = 16; m >= 1; m >>= 1) sum += __shfl_xor(sum, m, 32);
  if ((tid & 31) == 0) red[tid >> 5] = sum;
  __syncthreads();
  float tot = 0.f;
#pragma unroll
  for (int w = 0; w < 8; ++w) tot += red[w];
  const float inv = 1.0f / tot;
#pragma unroll
  for (int i = 0; i < LL / 256; ++i)
    attn[(size_t)b * LL + i * 256 + tid] = vals[i] * inv;
}

// ---------------------------------------------------------------------------
// Kernel 5: context[b,h] = sum_l attn[b,l] * keys[b,l,h]   (memory bound)
// ---------------------------------------------------------------------------
__global__ __launch_bounds__(HH) void context_kernel(const float* __restrict__ keys,
                                                     const float* __restrict__ attn,
                                                     float* __restrict__ ctx) {
  const int b = blockIdx.x;
  const int h = threadIdx.x;
  __shared__ float aw[256];
  float acc = 0.f;
  for (int l0 = 0; l0 < LL; l0 += 256) {
    __syncthreads();
    if (h < 256) aw[h] = attn[(size_t)b * LL + l0 + h];
    __syncthreads();
    const float* kp = keys + ((size_t)b * LL + l0) * HH + h;
#pragma unroll 8
    for (int l = 0; l < 256; ++l) acc += aw[l] * kp[(size_t)l * HH];
  }
  ctx[b * HH + h] = acc;
}

// ---------------------------------------------------------------------------
extern "C" void kernel_launch(void* const* d_in, const int* in_sizes, int n_in,
                              void* d_out, int out_size, void* d_ws, size_t ws_size,
                              hipStream_t stream) {
  (void)in_sizes; (void)n_in; (void)out_size; (void)ws_size;

  const float*         query = (const float*)d_in[0];
  const float*         keys  = (const float*)d_in[1];
  const unsigned char* mask  = (const unsigned char*)d_in[2];
  const float*         Wa_w  = (const float*)d_in[3];
  const float*         Wa_b  = (const float*)d_in[4];
  const float*         Ua_w  = (const float*)d_in[5];
  const float*         Ua_b  = (const float*)d_in[6];
  const float*         Va_w  = (const float*)d_in[7];
  const float*         Va_b  = (const float*)d_in[8];

  float* ctx  = (float*)d_out;        // (B, H)
  float* attn = ctx + BB * HH;        // (B, L)

  float*          qsum   = (float*)d_ws;                   // B*H f32
  unsigned short* Uab    = (unsigned short*)(qsum + BB * HH); // H*H bf16
  float*          scores = (float*)(Uab + HH * HH);        // B*L f32

  qproj_kernel<<<BB, HH, 0, stream>>>(query, Wa_w, Wa_b, Ua_b, qsum);
  cvt_bf16_kernel<<<(HH * HH) / 256, 256, 0, stream>>>(Ua_w, Uab, HH * HH);
  bahdanau_scores_kernel<<<dim3(BB, LL / 64), 128, 0, stream>>>(keys, Uab, qsum,
                                                                Va_w, Va_b, scores);
  softmax_kernel<<<BB, 256, 0, stream>>>(scores, mask, attn);
  context_kernel<<<BB, HH, 0, stream>>>(keys, attn, ctx);
}
